// MHA_42571715838425
// MI455X (gfx1250) — matrix operand order repaired and checked by reading, not compiled, not run on hardware
//
#include <hip/hip_runtime.h>
#include <hip/hip_bf16.h>

#define BATCH 4
#define SEQ   2048
#define CH    1024
#define NH    16
#define HD    64
#define C3    3072

typedef __attribute__((ext_vector_type(8)))  __bf16 bf16x8;
typedef __attribute__((ext_vector_type(16))) __bf16 bf16x16;
typedef __attribute__((ext_vector_type(8)))  float  f32x8;

__device__ __forceinline__ unsigned short f2bf(float f) {
    unsigned u = __builtin_bit_cast(unsigned, f);
    u += 0x7FFFu + ((u >> 16) & 1u);           // round-to-nearest-even
    return (unsigned short)(u >> 16);
}

__device__ __forceinline__ bf16x8 ld8(const unsigned short* p) {
    return *reinterpret_cast<const bf16x8*>(p);
}

__device__ __forceinline__ bf16x16 cat8(bf16x8 lo, bf16x8 hi) {
    bf16x16 r;
#pragma unroll
    for (int i = 0; i < 8; ++i) { r[i] = lo[i]; r[i + 8] = hi[i]; }
    return r;
}

__device__ __forceinline__ f32x8 zero8() {
    f32x8 z;
#pragma unroll
    for (int i = 0; i < 8; ++i) z[i] = 0.0f;
    return z;
}

__device__ __forceinline__ f32x8 wmma_bf16(bf16x16 a, bf16x16 b, f32x8 c) {
    return __builtin_amdgcn_wmma_f32_16x16x32_bf16(
        /*neg_a=*/false, a, /*neg_b=*/false, b,
        /*c_mod=*/(short)0, c, /*reuse_a=*/false, /*reuse_b=*/false);
}

// One k-step (32) worth of fragments for a 32x64 output tile.
struct FragSet { bf16x16 a0, a1, b[4]; };

__device__ __forceinline__ FragSet load_set(const unsigned short* ar0,
                                            const unsigned short* ar1,
                                            const unsigned short* const br[4],
                                            int k) {
    FragSet f;
    f.a0 = cat8(ld8(ar0 + k), ld8(ar0 + k + 16));
    f.a1 = cat8(ld8(ar1 + k), ld8(ar1 + k + 16));
#pragma unroll
    for (int t = 0; t < 4; ++t)
        f.b[t] = cat8(ld8(br[t] + k), ld8(br[t] + k + 16));
    return f;
}

__device__ __forceinline__ void mma_set(const FragSet& f, f32x8 acc[8]) {
#pragma unroll
    for (int t = 0; t < 4; ++t) {
        acc[t]     = wmma_bf16(f.a0, f.b[t], acc[t]);
        acc[4 + t] = wmma_bf16(f.a1, f.b[t], acc[4 + t]);
    }
}

// ---------------------------------------------------------------- fp32 -> bf16
__global__ void cvt_bf16_kernel(const float* __restrict__ src,
                                unsigned short* __restrict__ dst, int n) {
    int i = blockIdx.x * blockDim.x + threadIdx.x;
    if (i < n) dst[i] = f2bf(src[i]);
}

// ---------------------------------------------------------------- QKV GEMM
// A: x_bf16 [M=8192][K=1024]; Bw: Wqkv_bf16 [N=3072][K=1024] (both K-contiguous).
// Wave computes 32x64 tile; ping-pong double buffering (no register copies).
__global__ __launch_bounds__(256) void gemm_qkv_kernel(
    const unsigned short* __restrict__ A,
    const unsigned short* __restrict__ Bw,
    const float* __restrict__ bias,
    unsigned short* __restrict__ Qm,
    unsigned short* __restrict__ Km,
    unsigned short* __restrict__ Vt)
{
    const int K  = CH;
    const int Nt = C3 / 64;                     // 48
    const int Mt = BATCH * SEQ / 32;            // 256
    int lane = threadIdx.x & 31, wv = threadIdx.x >> 5;
    int tile = blockIdx.x * 8 + wv;
    if (tile >= Mt * Nt) return;
    int mt = tile / Nt, nt = tile % Nt;
    int ml = lane & 15, hi = lane >> 4;

    const unsigned short* ar0 = A + (size_t)(mt * 32 + ml) * K + 8 * hi;
    const unsigned short* ar1 = ar0 + (size_t)16 * K;
    const unsigned short* br[4];
#pragma unroll
    for (int t = 0; t < 4; ++t)
        br[t] = Bw + (size_t)(nt * 64 + t * 16 + ml) * K + 8 * hi;

    f32x8 acc[8];
#pragma unroll
    for (int i = 0; i < 8; ++i) acc[i] = zero8();

    FragSet f0 = load_set(ar0, ar1, br, 0);
    int k = 0;
    for (; k + 64 < K; k += 64) {
        FragSet f1 = load_set(ar0, ar1, br, k + 32);   // in flight over mma(f0)
        mma_set(f0, acc);
        f0 = load_set(ar0, ar1, br, k + 64);           // in flight over mma(f1)
        mma_set(f1, acc);
    }
    {   // tail: k == K - 64
        FragSet f1 = load_set(ar0, ar1, br, k + 32);
        mma_set(f0, acc);
        mma_set(f1, acc);
    }

#pragma unroll
    for (int mt2 = 0; mt2 < 2; ++mt2) {
#pragma unroll
        for (int t = 0; t < 4; ++t) {
            int n = nt * 64 + t * 16 + ml;
            float bv = bias[n];
#pragma unroll
            for (int r = 0; r < 8; ++r) {
                int m = mt * 32 + mt2 * 16 + r + 8 * hi;
                float v = acc[mt2 * 4 + t][r] + bv;
                int b = m >> 11, s = m & (SEQ - 1);
                unsigned short w = f2bf(v);
                if (n < CH) {
                    int h = n >> 6, d = n & 63;
                    Qm[(((size_t)(b * NH + h)) * SEQ + s) * HD + d] = w;
                } else if (n < 2 * CH) {
                    int c = n - CH; int h = c >> 6, d = c & 63;
                    Km[(((size_t)(b * NH + h)) * SEQ + s) * HD + d] = w;
                } else {
                    int c = n - 2 * CH; int h = c >> 6, d = c & 63;
                    Vt[(((size_t)(b * NH + h)) * HD + d) * SEQ + s] = w;
                }
            }
        }
    }
}

// ---------------------------------------------------------------- flash attention
// One wave per (b, h, 32-query tile). 4 waves / block, private LDS slices.
__global__ __launch_bounds__(128) void attn_kernel(
    const unsigned short* __restrict__ Qm,
    const unsigned short* __restrict__ Km,
    const unsigned short* __restrict__ Vt,
    unsigned short* __restrict__ Ob)
{
    __shared__ __align__(16) float          sS[4][32 * 32];
    __shared__ __align__(16) unsigned short sP[4][32 * 32];
    __shared__ float                        sA[4][32];

    int lane = threadIdx.x & 31, wv = threadIdx.x >> 5;
    int gw = blockIdx.x * 4 + wv;
    const int S32 = SEQ / 32;                   // 64
    int b   = gw / (NH * S32);
    int rem = gw % (NH * S32);
    int h   = rem / S32;
    int qt  = rem % S32;
    int q0  = qt * 32;
    int ml = lane & 15, hi = lane >> 4;

    const size_t headQK = ((size_t)(b * NH + h)) * SEQ * HD;
    const size_t headV  = ((size_t)(b * NH + h)) * HD * SEQ;

    // Q fragments: two 16-row tiles x two 16x32 A-matrices (d halves)
    bf16x16 qf[2][2];
#pragma unroll
    for (int mt = 0; mt < 2; ++mt) {
        const unsigned short* qrow =
            Qm + headQK + (size_t)(q0 + mt * 16 + ml) * HD + 8 * hi;
        qf[mt][0] = cat8(ld8(qrow),      ld8(qrow + 16));
        qf[mt][1] = cat8(ld8(qrow + 32), ld8(qrow + 48));
    }

    f32x8 o[2][4];
#pragma unroll
    for (int mt = 0; mt < 2; ++mt)
#pragma unroll
        for (int t = 0; t < 4; ++t) o[mt][t] = zero8();

    float mi = -3.0e38f, li = 0.0f;             // lane owns query row q0+lane
    const float scale = 0.125f;                 // 1/sqrt(64)

    // block-uniform trip count (4 waves share b,h; qt = 4k..4k+3)
    int kbu = (qt | 3);

    for (int kb = 0; kb <= kbu; ++kb) {
        // ---- S = Q K^T (32x32 block of keys) ----
#pragma unroll
        for (int kt = 0; kt < 2; ++kt) {
            int sk = kb * 32 + kt * 16;
            const unsigned short* krow =
                Km + headQK + (size_t)(sk + ml) * HD + 8 * hi;
            bf16x16 kf0 = cat8(ld8(krow),      ld8(krow + 16));
            bf16x16 kf1 = cat8(ld8(krow + 32), ld8(krow + 48));
#pragma unroll
            for (int mt = 0; mt < 2; ++mt) {
                f32x8 st = zero8();
                st = wmma_bf16(qf[mt][0], kf0, st);
                st = wmma_bf16(qf[mt][1], kf1, st);
#pragma unroll
                for (int r = 0; r < 8; ++r) {
                    int rowl = mt * 16 + r + 8 * hi;
                    int q  = q0 + rowl;
                    int kk = sk + ml;
                    float v = st[r] * scale;
                    if (kk > q) v = -1.0e30f;   // causal mask
                    sS[wv][rowl * 32 + kt * 16 + ml] = v;
                }
            }
        }

        // ---- prefetch V fragments (independent of softmax) ----
        bf16x16 vf[4];
#pragma unroll
        for (int t = 0; t < 4; ++t) {
            const unsigned short* vrow =
                Vt + headV + (size_t)(t * 16 + ml) * SEQ + kb * 32 + 8 * hi;
            vf[t] = cat8(ld8(vrow), ld8(vrow + 16));
        }
        __syncthreads();

        // ---- online softmax: each lane owns one of 32 rows ----
        {
            const float* row = &sS[wv][lane * 32];
            float bm = -3.0e38f;
#pragma unroll
            for (int k = 0; k < 32; ++k) bm = fmaxf(bm, row[k]);
            float nm    = fmaxf(mi, bm);
            float alpha = __expf(mi - nm);
            float sum = 0.0f;
#pragma unroll
            for (int k = 0; k < 32; ++k) {
                float p = __expf(row[k] - nm);
                sum += p;
                sP[wv][lane * 32 + k] = f2bf(p);
            }
            li = li * alpha + sum;
            mi = nm;
            sA[wv][lane] = alpha;
        }
        __syncthreads();

        // ---- O = diag(alpha) O + P V ----
#pragma unroll
        for (int mt = 0; mt < 2; ++mt) {
            float av[8];
#pragma unroll
            for (int r = 0; r < 8; ++r) av[r] = sA[wv][mt * 16 + r + 8 * hi];
            const unsigned short* prow = &sP[wv][(mt * 16 + ml) * 32 + 8 * hi];
            bf16x16 pf = cat8(ld8(prow), ld8(prow + 16));
#pragma unroll
            for (int t = 0; t < 4; ++t) {
#pragma unroll
                for (int r = 0; r < 8; ++r) o[mt][t][r] *= av[r];
                o[mt][t] = wmma_bf16(pf, vf[t], o[mt][t]);
            }
        }
        __syncthreads();
    }

    sA[wv][lane] = 1.0f / li;
    __syncthreads();

#pragma unroll
    for (int mt = 0; mt < 2; ++mt) {
        float inv[8];
#pragma unroll
        for (int r = 0; r < 8; ++r) inv[r] = sA[wv][mt * 16 + r + 8 * hi];
#pragma unroll
        for (int t = 0; t < 4; ++t) {
            int d = t * 16 + ml;
#pragma unroll
            for (int r = 0; r < 8; ++r) {
                int s = q0 + mt * 16 + r + 8 * hi;
                Ob[((size_t)(b * SEQ + s)) * CH + h * HD + d] =
                    f2bf(o[mt][t][r] * inv[r]);
            }
        }
    }
}

// ---------------------------------------------------------------- output projection
// Wave computes 32x64 tile; ping-pong double buffering; fp32 output + bias.
__global__ __launch_bounds__(256) void gemm_proj_kernel(
    const unsigned short* __restrict__ A,     // attn out bf16 [8192][1024]
    const unsigned short* __restrict__ Bw,    // Wproj bf16 [1024][1024]
    const float* __restrict__ bias,
    float* __restrict__ out)
{
    const int K  = CH;
    const int Nt = CH / 64;                    // 16
    const int Mt = BATCH * SEQ / 32;           // 256
    int lane = threadIdx.x & 31, wv = threadIdx.x >> 5;
    int tile = blockIdx.x * 8 + wv;
    if (tile >= Mt * Nt) return;
    int mt = tile / Nt, nt = tile % Nt;
    int ml = lane & 15, hi = lane >> 4;

    const unsigned short* ar0 = A + (size_t)(mt * 32 + ml) * K + 8 * hi;
    const unsigned short* ar1 = ar0 + (size_t)16 * K;
    const unsigned short* br[4];
#pragma unroll
    for (int t = 0; t < 4; ++t)
        br[t] = Bw + (size_t)(nt * 64 + t * 16 + ml) * K + 8 * hi;

    f32x8 acc[8];
#pragma unroll
    for (int i = 0; i < 8; ++i) acc[i] = zero8();

    FragSet f0 = load_set(ar0, ar1, br, 0);
    int k = 0;
    for (; k + 64 < K; k += 64) {
        FragSet f1 = load_set(ar0, ar1, br, k + 32);
        mma_set(f0, acc);
        f0 = load_set(ar0, ar1, br, k + 64);
        mma_set(f1, acc);
    }
    {   // tail: k == K - 64
        FragSet f1 = load_set(ar0, ar1, br, k + 32);
        mma_set(f0, acc);
        mma_set(f1, acc);
    }

#pragma unroll
    for (int mt2 = 0; mt2 < 2; ++mt2) {
#pragma unroll
        for (int t = 0; t < 4; ++t) {
            int n = nt * 64 + t * 16 + ml;
            float bv = bias[n];
#pragma unroll
            for (int r = 0; r < 8; ++r) {
                int m = mt * 32 + mt2 * 16 + r + 8 * hi;
                out[(size_t)m * CH + n] = acc[mt2 * 4 + t][r] + bv;
            }
        }
    }
}

// ---------------------------------------------------------------- host launcher
extern "C" void kernel_launch(void* const* d_in, const int* in_sizes, int n_in,
                              void* d_out, int out_size, void* d_ws, size_t ws_size,
                              hipStream_t stream) {
    const float* x     = (const float*)d_in[0];
    const float* Wqkv  = (const float*)d_in[1];
    const float* bqkv  = (const float*)d_in[2];
    const float* Wproj = (const float*)d_in[3];
    const float* bproj = (const float*)d_in[4];
    float* out = (float*)d_out;

    char* ws = (char*)d_ws;
    const size_t MB = 1024 * 1024;
    unsigned short* xb     = (unsigned short*)(ws);              // 16 MB
    unsigned short* wqkvb  = (unsigned short*)(ws + 16 * MB);    //  6 MB
    unsigned short* wprojb = (unsigned short*)(ws + 22 * MB);    //  2 MB
    unsigned short* Qm     = (unsigned short*)(ws + 24 * MB);    // 16 MB
    unsigned short* Km     = (unsigned short*)(ws + 40 * MB);    // 16 MB
    unsigned short* Vt     = (unsigned short*)(ws + 56 * MB);    // 16 MB
    unsigned short* attnb  = (unsigned short*)(ws + 72 * MB);    // 16 MB  (total 88 MB)

    const int nx  = BATCH * SEQ * CH;   // 8388608
    const int nwq = C3 * CH;            // 3145728
    const int nwp = CH * CH;            // 1048576

    cvt_bf16_kernel<<<(nx  + 255) / 256, 256, 0, stream>>>(x,     xb,     nx);
    cvt_bf16_kernel<<<(nwq + 255) / 256, 256, 0, stream>>>(Wqkv,  wqkvb,  nwq);
    cvt_bf16_kernel<<<(nwp + 255) / 256, 256, 0, stream>>>(Wproj, wprojb, nwp);

    // (8192/32)*(3072/64) = 12288 wave-tiles / 8 waves per block
    gemm_qkv_kernel<<<1536, 256, 0, stream>>>(xb, wqkvb, bqkv, Qm, Km, Vt);

    // 4*16*64 = 4096 q-tiles(32) / 4 waves per block
    attn_kernel<<<1024, 128, 0, stream>>>(Qm, Km, Vt, attnb);

    // (8192/32)*(1024/64) = 4096 wave-tiles / 8 waves per block
    gemm_proj_kernel<<<512, 256, 0, stream>>>(attnb, wprojb, bproj, out);
}
